// NTXentLoss_57200374448738
// MI455X (gfx1250) — matrix operand order repaired and checked by reading, not compiled, not run on hardware
//
#include <hip/hip_runtime.h>
#include <hip/hip_bf16.h>

typedef __attribute__((ext_vector_type(16))) _Float16 v16h;
typedef __attribute__((ext_vector_type(8)))  _Float16 v8h;
typedef __attribute__((ext_vector_type(8)))  float    v8f;
typedef __attribute__((ext_vector_type(4)))  float    f32x4;

#define BATCH 4096
#define DIM   256
#define ROWS2 8192
#define NEGK  10
#define INV_T 2.0f   // 1 / TEMP, TEMP = 0.5

union Frag {
  v16h h;
  struct { f32x4 lo, hi; } q;
};

// Insert v into descending-sorted 10-list (caller guarantees v > L[9]).
__device__ __forceinline__ void insert10(float (&L)[NEGK], float v) {
  #pragma unroll
  for (int j = NEGK - 1; j >= 1; --j)
    L[j] = (v > L[j - 1]) ? L[j - 1] : ((v > L[j]) ? v : L[j]);
  L[0] = fmaxf(L[0], v);
}

// Kernel 1: per batch-row r, normalize z_i[r] and z_j[r], emit f16 rows
// N[r] and N[r+B], positive pos[r] = <n_i, n_j>, and zero the output.
__global__ __launch_bounds__(32) void ntx_norm(const float* __restrict__ zi,
                                               const float* __restrict__ zj,
                                               _Float16* __restrict__ Nh,
                                               float* __restrict__ posArr,
                                               float* __restrict__ out) {
  const int r    = blockIdx.x;
  const int lane = threadIdx.x;          // 32 lanes, 8 elems each
  const float* xi = zi + (size_t)r * DIM + lane * 8;
  const float* xj = zj + (size_t)r * DIM + lane * 8;
  f32x4 ia0 = *(const f32x4*)(xi);
  f32x4 ia1 = *(const f32x4*)(xi + 4);
  f32x4 ja0 = *(const f32x4*)(xj);
  f32x4 ja1 = *(const f32x4*)(xj + 4);
  float si = 0.f, sj = 0.f, dp = 0.f;
  #pragma unroll
  for (int e = 0; e < 4; ++e) {
    si += ia0[e] * ia0[e] + ia1[e] * ia1[e];
    sj += ja0[e] * ja0[e] + ja1[e] * ja1[e];
    dp += ia0[e] * ja0[e] + ia1[e] * ja1[e];
  }
  #pragma unroll
  for (int off = 16; off >= 1; off >>= 1) {   // wave32 butterfly
    si += __shfl_xor(si, off);
    sj += __shfl_xor(sj, off);
    dp += __shfl_xor(dp, off);
  }
  float yi = rsqrtf(si); yi = yi * (1.5f - 0.5f * si * yi * yi);  // Newton step
  float yj = rsqrtf(sj); yj = yj * (1.5f - 0.5f * sj * yj * yj);
  if (lane == 0) {
    posArr[r] = dp * yi * yj;
    if (r == 0) out[0] = 0.0f;   // kernel boundary orders this before atomics
  }
  v8h oi, oj;
  #pragma unroll
  for (int e = 0; e < 4; ++e) {
    oi[e]     = (_Float16)(ia0[e] * yi);
    oi[e + 4] = (_Float16)(ia1[e] * yi);
    oj[e]     = (_Float16)(ja0[e] * yj);
    oj[e + 4] = (_Float16)(ja1[e] * yj);
  }
  *(v8h*)(Nh + (size_t)r * DIM + lane * 8)           = oi;
  *(v8h*)(Nh + (size_t)(r + BATCH) * DIM + lane * 8) = oj;
}

// Kernel 2: fused sim-GEMM + per-row top-10 + loss.
// 4 waves/block; each wave owns 16 rows, sweeps all 8192 columns.
__global__ __launch_bounds__(128) void ntx_topk(const _Float16* __restrict__ Nh,
                                                const float* __restrict__ posArr,
                                                float* __restrict__ out) {
  __shared__ float smem[4 * 16 * 16 * NEGK];   // 40 KB: wave x row x col-lane x K
  const int lane = threadIdx.x & 31;
  const int w    = threadIdx.x >> 5;
  const int gw   = blockIdx.x * 4 + w;
  const int R    = gw * 16;                    // first row of this wave's block
  const int hi   = lane >> 4;
  const int lo   = lane & 15;

  // Preload A fragments (rows R..R+15, all K=0..255): 8 x v16h, reused all sweep.
  // Layout (16-bit A 16x32): lanes0-15 row=lane K{0..7,16..23}; lanes16-31 K{8..15,24..31}.
  const _Float16* aptr = Nh + (size_t)(R + lo) * DIM + 8 * hi;
  Frag A[8];
  #pragma unroll
  for (int k = 0; k < 8; ++k) {
    A[k].q.lo = *(const f32x4*)(aptr + 32 * k);
    A[k].q.hi = *(const f32x4*)(aptr + 32 * k + 16);
  }

  // Per-lane top-10 lists for 8 rows (C layout: VGPR r -> row r + 8*hi, col = lo).
  float L[8][NEGK];
  #pragma unroll
  for (int r = 0; r < 8; ++r)
    #pragma unroll
    for (int j = 0; j < NEGK; ++j) L[r][j] = -1e30f;

  // Only two 16-col tiles per row-block contain masked (self/positive) columns.
  const int td0   = (R & (BATCH - 1)) >> 4;
  const int td1   = td0 + (BATCH >> 4);
  const int dslot = (lane < 16) ? lane : (lane - 24);  // diag VGPR slot, valid 0..7

  #pragma unroll 1
  for (int t = 0; t < ROWS2 / 16; ++t) {
    // B layout (16-bit B 32x16): lane col = lo; lanes0-15 K{0..7,16..23},
    // lanes16-31 K{8..15,24..31} (matches A).
    const _Float16* bptr = Nh + (size_t)(t * 16 + lo) * DIM + 8 * hi;
    v8f c = {};
    #pragma unroll
    for (int k = 0; k < 8; ++k) {
      Frag B;
      B.q.lo = *(const f32x4*)(bptr + 32 * k);
      B.q.hi = *(const f32x4*)(bptr + 32 * k + 16);
      c = __builtin_amdgcn_wmma_f32_16x16x32_f16(false, A[k].h, false, B.h,
                                                 (short)0, c, false, false);
    }
    float v[8];
    #pragma unroll
    for (int r = 0; r < 8; ++r) v[r] = c[r];
    if (t == td0 || t == td1) {      // mask col % B == row % B (self + positive)
      #pragma unroll
      for (int r = 0; r < 8; ++r)
        if (dslot == r) v[r] = -1e30f;
    }
    #pragma unroll
    for (int r = 0; r < 8; ++r)
      if (v[r] > L[r][NEGK - 1]) insert10(L[r], v[r]);   // rare after warmup
  }

  // Dump per-lane lists to LDS: [row 0..15][col-lane 0..15][10], sorted desc.
  float* base = smem + w * (16 * 16 * NEGK);
  #pragma unroll
  for (int r = 0; r < 8; ++r) {
    const int m = r + 8 * hi;
    #pragma unroll
    for (int j = 0; j < NEGK; ++j)
      base[(m * 16 + lo) * NEGK + j] = L[r][j];
  }
  __syncthreads();

  // Merge 16 sorted lists per row (lanes 0..15, one row each), then the loss.
  if (lane < 16) {
    float M[NEGK];
    #pragma unroll
    for (int j = 0; j < NEGK; ++j) M[j] = -1e30f;
    const int m = lane;
    for (int cc = 0; cc < 16; ++cc) {
      const float* q = base + (m * 16 + cc) * NEGK;
      for (int j = 0; j < NEGK; ++j) {
        float v = q[j];
        if (v <= M[NEGK - 1]) break;   // source sorted desc -> early exit
        insert10(M, v);
      }
    }
    const int row = R + m;
    const float p  = posArr[row & (BATCH - 1)];
    const float mx = fmaxf(p, M[0]);
    float s = __expf((p - mx) * INV_T);
    #pragma unroll
    for (int j = 0; j < NEGK; ++j) s += __expf((M[j] - mx) * INV_T);
    const float loss = __logf(s) + (mx - p) * INV_T;   // = LSE - pos/T
    atomicAdd(out, loss * (1.0f / (float)ROWS2));
  }
}

extern "C" void kernel_launch(void* const* d_in, const int* in_sizes, int n_in,
                              void* d_out, int out_size, void* d_ws, size_t ws_size,
                              hipStream_t stream) {
  (void)in_sizes; (void)n_in; (void)out_size; (void)ws_size;
  const float* zi = (const float*)d_in[0];
  const float* zj = (const float*)d_in[1];
  float* out = (float*)d_out;
  _Float16* Nh   = (_Float16*)d_ws;                                   // 4 MB
  float*    posA = (float*)((char*)d_ws + (size_t)ROWS2 * DIM * 2);   // +16 KB
  ntx_norm<<<BATCH, 32, 0, stream>>>(zi, zj, Nh, posA, out);
  ntx_topk<<<ROWS2 / 64, 128, 0, stream>>>(Nh, posA, out);
}
